// ParallelTransformerBlock_20255065768468
// MI455X (gfx1250) — compile-verified, matrix operand order checked by reading, not run on hardware
//
#include <hip/hip_runtime.h>

#define N_TOK 2048
#define DIM 2048
#define HEADS 16
#define DH 128
#define ATTN_INNER 2048
#define FF_INNER 8192
#define FUSED_OUT 18656          // 2048 + 256 + 16384
#define KOFF 2048                // k column offset in proj
#define VOFF 2176                // v column offset in proj
#define FFOFF 2304               // ff column offset in proj

typedef __attribute__((ext_vector_type(16))) __bf16 bf16x16;
typedef __attribute__((ext_vector_type(8)))  float  f32x8;
typedef __attribute__((ext_vector_type(4)))  int    v4i_;
typedef __attribute__((address_space(1)))    v4i_   gv4i;   // global int4
typedef __attribute__((address_space(3)))    v4i_   lv4i;   // LDS int4

union FragB16 { bf16x16 v; uint4 u[2]; };

#if __has_builtin(__builtin_amdgcn_global_load_async_to_lds_b128) && \
    __has_builtin(__builtin_amdgcn_s_wait_asynccnt)
#define HAVE_ASYNC_LDS 1
#else
#define HAVE_ASYNC_LDS 0
#endif

__device__ __forceinline__ unsigned short f2bf(float f) {
    union { float f; unsigned u; } v; v.f = f;
    unsigned r = v.u + 0x7FFFu + ((v.u >> 16) & 1u);   // round-to-nearest-even
    return (unsigned short)(r >> 16);
}
__device__ __forceinline__ float bf2f(unsigned short h) {
    union { unsigned u; float f; } v; v.u = ((unsigned)h) << 16;
    return v.f;
}

// Stage one 16-byte chunk global -> LDS (async on CDNA5, sync fallback).
__device__ __forceinline__ void stage16(unsigned short* ldsp, const unsigned short* gp) {
#if HAVE_ASYNC_LDS
    __builtin_amdgcn_global_load_async_to_lds_b128(
        (gv4i*)(uintptr_t)gp,
        (lv4i*)(unsigned)(uintptr_t)ldsp,
        0, 0);
#else
    *reinterpret_cast<uint4*>(ldsp) = *reinterpret_cast<const uint4*>(gp);
#endif
}
__device__ __forceinline__ void stage_wait() {
#if HAVE_ASYNC_LDS
    __builtin_amdgcn_s_wait_asynccnt(0);
#endif
}

// ---------------------------------------------------------------- LayerNorm
__global__ __launch_bounds__(256)
void ln_kernel(const float* __restrict__ x, const float* __restrict__ gamma,
               unsigned short* __restrict__ xn) {
    const int row = blockIdx.x;
    const float* xr = x + (size_t)row * DIM;
    __shared__ float red0[256], red1[256];
    float s = 0.f, ss = 0.f;
    for (int c = threadIdx.x; c < DIM; c += 256) { float v = xr[c]; s += v; ss += v * v; }
    red0[threadIdx.x] = s; red1[threadIdx.x] = ss;
    __syncthreads();
    for (int off = 128; off > 0; off >>= 1) {
        if (threadIdx.x < off) {
            red0[threadIdx.x] += red0[threadIdx.x + off];
            red1[threadIdx.x] += red1[threadIdx.x + off];
        }
        __syncthreads();
    }
    const float mu  = red0[0] * (1.f / DIM);
    const float var = red1[0] * (1.f / DIM) - mu * mu;
    const float rs  = rsqrtf(var + 1e-5f);
    for (int c = threadIdx.x; c < DIM; c += 256)
        xn[(size_t)row * DIM + c] = f2bf((xr[c] - mu) * rs * gamma[c]);
}

// ---------------------------------------------------------------- f32 [K][N] -> bf16 dst[N][K]
__global__ __launch_bounds__(256)
void transpose_cast_bf16(const float* __restrict__ src, unsigned short* __restrict__ dst,
                         int K, int N) {
    __shared__ unsigned short tile[32][33];
    const int kb = blockIdx.y * 32;
    const int nb = blockIdx.x * 32;
    const int tc = threadIdx.x & 31;
    const int tr = threadIdx.x >> 5;
#pragma unroll
    for (int i = 0; i < 4; ++i)
        tile[tr + i * 8][tc] = f2bf(src[(size_t)(kb + tr + i * 8) * N + nb + tc]);
    __syncthreads();
#pragma unroll
    for (int i = 0; i < 4; ++i)
        dst[(size_t)(nb + tr + i * 8) * K + kb + tc] = tile[tc][tr + i * 8];
}

// ---------------------------------------------------------------- bf16 [R][C] (row stride ld) -> dst[C][R]
__global__ __launch_bounds__(256)
void transpose_bf16(const unsigned short* __restrict__ src, int ld,
                    unsigned short* __restrict__ dst, int R, int C) {
    __shared__ unsigned short tile[32][33];
    const int rb = blockIdx.y * 32;
    const int cb = blockIdx.x * 32;
    const int tc = threadIdx.x & 31;
    const int tr = threadIdx.x >> 5;
#pragma unroll
    for (int i = 0; i < 4; ++i)
        tile[tr + i * 8][tc] = src[(size_t)(rb + tr + i * 8) * ld + cb + tc];
    __syncthreads();
#pragma unroll
    for (int i = 0; i < 4; ++i)
        dst[(size_t)(cb + tr + i * 8) * R + rb + tc] = tile[tc][tr + i * 8];
}

// ---------------------------------------------------------------- GEMM: C[M,N] = A[M,K] @ Bt[N,K]^T
// Both operands K-major -> symmetric b128 LDS staging, async double-buffered.
template <bool OUT_BF16, bool ACC>
__global__ __launch_bounds__(256)
void gemm_tt(const unsigned short* __restrict__ A,   // [M][K]
             const unsigned short* __restrict__ Bt,  // [N][K]
             void* __restrict__ C, int M, int N, int K) {
    __shared__ unsigned short As[2][128][32];
    __shared__ unsigned short Bs[2][128][32];

    const int bm = blockIdx.y * 128;
    const int bn = blockIdx.x * 128;
    const int tid  = threadIdx.x;
    const int lane = tid & 31;
    const int wave = tid >> 5;
    const int wm = (wave >> 2) * 64;
    const int wn = (wave & 3) * 32;
    const int lo = (lane >> 4) * 8;
    const int lr = lane & 15;

    f32x8 acc[4][2];
#pragma unroll
    for (int i = 0; i < 4; ++i)
#pragma unroll
        for (int j = 0; j < 2; ++j)
#pragma unroll
            for (int e = 0; e < 8; ++e) acc[i][j][e] = 0.f;

    const int srow = tid >> 2;           // 0..63
    const int skc  = (tid & 3) * 8;      // 0/8/16/24

    auto stage_tiles = [&](int kt, int buf) {
#pragma unroll
        for (int c = 0; c < 2; ++c) {
            int r = srow + c * 64;
            stage16(&As[buf][r][skc], A + (size_t)(bm + r) * K + kt + skc);
            int gn = bn + r; if (gn > N - 1) gn = N - 1;
            stage16(&Bs[buf][r][skc], Bt + (size_t)gn * K + kt + skc);
        }
    };

    const int nk = K / 32;
    stage_tiles(0, 0);
    for (int t = 0; t < nk; ++t) {
        const int buf = t & 1;
        stage_wait();
        __syncthreads();
        if (t + 1 < nk) stage_tiles((t + 1) * 32, buf ^ 1);

        FragB16 a[4], b[2];
#pragma unroll
        for (int i = 0; i < 4; ++i) {
            int r = wm + i * 16 + lr;
            a[i].u[0] = *reinterpret_cast<const uint4*>(&As[buf][r][lo]);
            a[i].u[1] = *reinterpret_cast<const uint4*>(&As[buf][r][lo + 16]);
        }
#pragma unroll
        for (int j = 0; j < 2; ++j) {
            int n = wn + j * 16 + lr;
            b[j].u[0] = *reinterpret_cast<const uint4*>(&Bs[buf][n][lo]);
            b[j].u[1] = *reinterpret_cast<const uint4*>(&Bs[buf][n][lo + 16]);
        }
#pragma unroll
        for (int i = 0; i < 4; ++i)
#pragma unroll
            for (int j = 0; j < 2; ++j)
                acc[i][j] = __builtin_amdgcn_wmma_f32_16x16x32_bf16(
                    false, a[i].v, false, b[j].v, (short)0, acc[i][j], false, false);
    }

    const int half = lane >> 4;
#pragma unroll
    for (int i = 0; i < 4; ++i)
#pragma unroll
        for (int j = 0; j < 2; ++j)
#pragma unroll
            for (int r = 0; r < 8; ++r) {
                int gm = bm + wm + i * 16 + r + half * 8;
                int gn = bn + wn + j * 16 + lr;
                if (gn < N) {
                    size_t idx = (size_t)gm * N + gn;
                    if (OUT_BF16) {
                        ((unsigned short*)C)[idx] = f2bf(acc[i][j][r]);
                    } else {
                        float* co = (float*)C;
                        co[idx] = ACC ? (co[idx] + acc[i][j][r]) : acc[i][j][r];
                    }
                }
            }
}

// ---------------------------------------------------------------- elementwise prep
__global__ __launch_bounds__(256)
void prep_q(const unsigned short* __restrict__ proj, unsigned short* __restrict__ qs) {
    size_t i = (size_t)blockIdx.x * 256 + threadIdx.x;
    int row = (int)(i >> 11);
    int c   = (int)(i & 2047);
    float v = bf2f(proj[(size_t)row * FUSED_OUT + c]) * 0.08838834764831845f; // 128^-0.5
    qs[i] = f2bf(v);
}

__global__ __launch_bounds__(256)
void prep_ff(const unsigned short* __restrict__ proj, unsigned short* __restrict__ ffact) {
    size_t i = (size_t)blockIdx.x * 256 + threadIdx.x;
    int row = (int)(i >> 13);
    int c   = (int)(i & 8191);
    const unsigned short* pr = proj + (size_t)row * FUSED_OUT + FFOFF;
    float fx = bf2f(pr[c]);
    float g  = bf2f(pr[FF_INNER + c]);
    float sg = g / (1.f + __expf(-g));                   // silu(gate)
    ffact[i] = f2bf(sg * fx);
}

// ---------------------------------------------------------------- flash attention (multi-query)
// grid: (N_TOK/64, HEADS), block 128 (4 waves, 16 q-rows each). Barrier-free inner loop:
// K fragments read direct from proj (columns of K^T are contiguous K rows),
// V fragments read direct from pre-transposed vT[feat][key].
__global__ __launch_bounds__(128)
void attn_kernel(const unsigned short* __restrict__ qs,    // [N, 2048] scaled bf16
                 const unsigned short* __restrict__ proj,  // [N, 18656] (k slice)
                 const unsigned short* __restrict__ vT,    // [128, N]
                 unsigned short* __restrict__ obuf) {      // [N, 2048]
    const int h    = blockIdx.y;
    const int tid  = threadIdx.x;
    const int lane = tid & 31;
    const int wave = tid >> 5;
    const int lo   = (lane >> 4) * 8;
    const int lr   = lane & 15;
    const int half = lane >> 4;
    const int q0   = blockIdx.x * 64 + wave * 16;

    __shared__ unsigned short Pl[4][16][32];     // per-wave P staging (no barriers needed)

    FragB16 qf[4];
    {
        const unsigned short* qrow = qs + (size_t)(q0 + lr) * ATTN_INNER + h * DH;
#pragma unroll
        for (int c = 0; c < 4; ++c) {
            qf[c].u[0] = *reinterpret_cast<const uint4*>(qrow + c * 32 + lo);
            qf[c].u[1] = *reinterpret_cast<const uint4*>(qrow + c * 32 + lo + 16);
        }
    }

    f32x8 o[8];
#pragma unroll
    for (int f = 0; f < 8; ++f)
#pragma unroll
        for (int e = 0; e < 8; ++e) o[f][e] = 0.f;
    float m[8], l[8];
#pragma unroll
    for (int r = 0; r < 8; ++r) { m[r] = -1e30f; l[r] = 0.f; }

    const int jend = q0 + 16;                    // per-wave causal limit
    for (int j = 0; j < jend; j += 32) {
        // S = Q @ K^T (two 16-key tiles)
        f32x8 s0, s1;
#pragma unroll
        for (int e = 0; e < 8; ++e) { s0[e] = 0.f; s1[e] = 0.f; }
#pragma unroll
        for (int jt = 0; jt < 2; ++jt) {
            const unsigned short* krow =
                proj + (size_t)(j + jt * 16 + lr) * FUSED_OUT + KOFF;
#pragma unroll
            for (int c = 0; c < 4; ++c) {
                FragB16 kf;
                kf.u[0] = *reinterpret_cast<const uint4*>(krow + c * 32 + lo);
                kf.u[1] = *reinterpret_cast<const uint4*>(krow + c * 32 + lo + 16);
                if (jt == 0)
                    s0 = __builtin_amdgcn_wmma_f32_16x16x32_bf16(
                        false, qf[c].v, false, kf.v, (short)0, s0, false, false);
                else
                    s1 = __builtin_amdgcn_wmma_f32_16x16x32_bf16(
                        false, qf[c].v, false, kf.v, (short)0, s1, false, false);
            }
        }

        // causal mask + online softmax
#pragma unroll
        for (int r = 0; r < 8; ++r) {
            int row = q0 + r + half * 8;
            if (j + lr      > row) s0[r] = -1e30f;
            if (j + 16 + lr > row) s1[r] = -1e30f;
        }
#pragma unroll
        for (int r = 0; r < 8; ++r) {
            float mx = fmaxf(s0[r], s1[r]);
#pragma unroll
            for (int d = 8; d >= 1; d >>= 1) mx = fmaxf(mx, __shfl_xor(mx, d, 16));
            float mnew  = fmaxf(m[r], mx);
            float scale = __expf(m[r] - mnew);
            float p0 = __expf(s0[r] - mnew);
            float p1 = __expf(s1[r] - mnew);
            float rs = p0 + p1;
#pragma unroll
            for (int d = 8; d >= 1; d >>= 1) rs += __shfl_xor(rs, d, 16);
            l[r] = l[r] * scale + rs;
            m[r] = mnew;
#pragma unroll
            for (int f = 0; f < 8; ++f) o[f][r] *= scale;
            Pl[wave][r + half * 8][lr]      = f2bf(p0);
            Pl[wave][r + half * 8][16 + lr] = f2bf(p1);
        }

        // O += P(16x32) @ V(32x128), V fragments straight from vT
        FragB16 pf;
        pf.u[0] = *reinterpret_cast<const uint4*>(&Pl[wave][lr][lo]);
        pf.u[1] = *reinterpret_cast<const uint4*>(&Pl[wave][lr][lo + 16]);
#pragma unroll
        for (int f = 0; f < 8; ++f) {
            const unsigned short* vrow = vT + (size_t)(f * 16 + lr) * N_TOK + j;
            FragB16 vf;
            vf.u[0] = *reinterpret_cast<const uint4*>(vrow + lo);
            vf.u[1] = *reinterpret_cast<const uint4*>(vrow + lo + 16);
            o[f] = __builtin_amdgcn_wmma_f32_16x16x32_bf16(
                false, pf.v, false, vf.v, (short)0, o[f], false, false);
        }
    }

    // epilogue
#pragma unroll
    for (int r = 0; r < 8; ++r) {
        float inv = 1.f / l[r];
        int row = q0 + r + half * 8;
#pragma unroll
        for (int f = 0; f < 8; ++f)
            obuf[(size_t)row * ATTN_INNER + h * DH + f * 16 + lr] =
                f2bf(o[f][r] * inv);
    }
}

// ---------------------------------------------------------------- launcher
extern "C" void kernel_launch(void* const* d_in, const int* in_sizes, int n_in,
                              void* d_out, int out_size, void* d_ws, size_t ws_size,
                              hipStream_t stream) {
    const float* x          = (const float*)d_in[0];
    const float* gamma      = (const float*)d_in[1];
    const float* w_fused    = (const float*)d_in[2];
    const float* w_attn_out = (const float*)d_in[3];
    const float* w_ff_out   = (const float*)d_in[4];
    float* out = (float*)d_out;

    char* ws = (char*)d_ws;
    unsigned short* xn    = (unsigned short*)ws; ws += (size_t)N_TOK * DIM * 2;
    unsigned short* wfT   = (unsigned short*)ws; ws += (size_t)DIM * FUSED_OUT * 2;   // [18656][2048]
    unsigned short* waoT  = (unsigned short*)ws; ws += (size_t)ATTN_INNER * DIM * 2;  // [2048][2048]
    unsigned short* wfoT  = (unsigned short*)ws; ws += (size_t)FF_INNER * DIM * 2;    // [2048][8192]
    unsigned short* proj  = (unsigned short*)ws; ws += (size_t)N_TOK * FUSED_OUT * 2;
    unsigned short* qsb   = (unsigned short*)ws; ws += (size_t)N_TOK * ATTN_INNER * 2;
    unsigned short* ffact = (unsigned short*)ws; ws += (size_t)N_TOK * FF_INNER * 2;
    unsigned short* obuf  = (unsigned short*)ws; ws += (size_t)N_TOK * ATTN_INNER * 2;
    unsigned short* vT    = (unsigned short*)ws; ws += (size_t)DH * N_TOK * 2;        // [128][2048]

    // 1. LayerNorm + transposed weight casts
    ln_kernel<<<N_TOK, 256, 0, stream>>>(x, gamma, xn);
    transpose_cast_bf16<<<dim3(FUSED_OUT / 32, DIM / 32), 256, 0, stream>>>(
        w_fused, wfT, DIM, FUSED_OUT);
    transpose_cast_bf16<<<dim3(DIM / 32, ATTN_INNER / 32), 256, 0, stream>>>(
        w_attn_out, waoT, ATTN_INNER, DIM);
    transpose_cast_bf16<<<dim3(DIM / 32, FF_INNER / 32), 256, 0, stream>>>(
        w_ff_out, wfoT, FF_INNER, DIM);

    // 2. fused projection: [2048,2048] @ [2048,18656] -> bf16
    dim3 g1((FUSED_OUT + 127) / 128, N_TOK / 128);
    gemm_tt<true, false><<<g1, 256, 0, stream>>>(xn, wfT, proj, N_TOK, FUSED_OUT, DIM);

    // 3. prep: q scaling, SwiGLU, V transpose
    prep_q<<<(N_TOK * ATTN_INNER) / 256, 256, 0, stream>>>(proj, qsb);
    prep_ff<<<(N_TOK * FF_INNER) / 256, 256, 0, stream>>>(proj, ffact);
    transpose_bf16<<<dim3(DH / 32, N_TOK / 32), 256, 0, stream>>>(
        proj + VOFF, FUSED_OUT, vT, N_TOK, DH);

    // 4. flash attention (multi-query, causal)
    attn_kernel<<<dim3(N_TOK / 64, HEADS), 128, 0, stream>>>(qsb, proj, vT, obuf);

    // 5. output projections into fp32 d_out (second accumulates)
    dim3 g2(DIM / 128, N_TOK / 128);
    gemm_tt<false, false><<<g2, 256, 0, stream>>>(obuf, waoT, out, N_TOK, DIM, ATTN_INNER);
    gemm_tt<false, true ><<<g2, 256, 0, stream>>>(ffact, wfoT, out, N_TOK, DIM, FF_INNER);
}
